// SoftGatingMoE_730144440862
// MI455X (gfx1250) — compile-verified, matrix operand order checked by reading, not run on hardware
//
#include <hip/hip_runtime.h>
#include <hip/hip_bf16.h>

// Problem constants (match reference)
#define NTOK 4096   // B*S
#define DDIM 1024
#define HDIM 2048
#define NEXP 8
#define BB 4
#define SS 1024

typedef __attribute__((ext_vector_type(16))) __bf16 v16bf;
typedef __attribute__((ext_vector_type(8)))  __bf16 v8bf;
typedef __attribute__((ext_vector_type(4)))  __bf16 v4bf;
typedef __attribute__((ext_vector_type(8)))  float  v8f;

__device__ __forceinline__ v8f vzero8() {
  v8f r;
#pragma unroll
  for (int i = 0; i < 8; ++i) r[i] = 0.f;
  return r;
}

// Assemble a 16-element bf16 operand fragment from two contiguous 16B runs.
// Per the CDNA5 bf16 16x16x32 layout, each lane holds K = {koff..koff+7,
// 16+koff..16+koff+7} of its row/column, which are two contiguous 8-element
// (16-byte) runs in row-major memory.
__device__ __forceinline__ v16bf ld16bf(const __bf16* p) {
  v8bf lo = *(const v8bf*)(p);
  v8bf hi = *(const v8bf*)(p + 16);
  return __builtin_shufflevector(lo, hi, 0, 1, 2, 3, 4, 5, 6, 7,
                                 8, 9, 10, 11, 12, 13, 14, 15);
}

__device__ __forceinline__ v8f wmma_bf16(v16bf a, v16bf b, v8f c) {
  return __builtin_amdgcn_wmma_f32_16x16x32_bf16(false, a, false, b,
                                                 (short)0, c, false, false);
}

// ---------------- utility kernels ----------------

__global__ void zero_f32(float* __restrict__ p, int n4) {
  int i = blockIdx.x * blockDim.x + threadIdx.x;
  if (i < n4) ((float4*)p)[i] = make_float4(0.f, 0.f, 0.f, 0.f);
}

__global__ void cvt_bf16(const float* __restrict__ s, __bf16* __restrict__ d,
                         int n4) {
  int i = blockIdx.x * blockDim.x + threadIdx.x;
  if (i < n4) {
    float4 f = ((const float4*)s)[i];
    v4bf v;
    v[0] = (__bf16)f.x; v[1] = (__bf16)f.y;
    v[2] = (__bf16)f.z; v[3] = (__bf16)f.w;
    *(v4bf*)(d + 4 * (size_t)i) = v;
  }
}

// ---------------- gating: logits, softmax, top-2, expert lists ----------------

__global__ void __launch_bounds__(256) gate_kernel(
    const float* __restrict__ x, const float* __restrict__ Wg,
    float* __restrict__ gate_out, float* __restrict__ rout_out,
    float* __restrict__ combined, int* __restrict__ counts,
    int* __restrict__ lists) {
  __shared__ float wg[NEXP * DDIM];  // 32 KB
  int tid = threadIdx.x;
  for (int i = tid; i < NEXP * DDIM; i += 256) wg[i] = Wg[i];
  __syncthreads();

  int lane = tid & 31;
  int n = blockIdx.x * 8 + (tid >> 5);  // one wave per token

  float acc[NEXP];
#pragma unroll
  for (int e = 0; e < NEXP; ++e) acc[e] = 0.f;
  const float* xr = x + (size_t)n * DDIM;
  for (int d = lane; d < DDIM; d += 32) {
    float xv = xr[d];
#pragma unroll
    for (int e = 0; e < NEXP; ++e) acc[e] += xv * wg[e * DDIM + d];
  }
#pragma unroll
  for (int off = 16; off > 0; off >>= 1) {
#pragma unroll
    for (int e = 0; e < NEXP; ++e) acc[e] += __shfl_xor(acc[e], off, 32);
  }

  if (lane == 0) {
    float m = acc[0];
    for (int e = 1; e < NEXP; ++e) m = fmaxf(m, acc[e]);
    float p[NEXP], s = 0.f;
    for (int e = 0; e < NEXP; ++e) { p[e] = __expf(acc[e] - m); s += p[e]; }
    float inv = 1.f / s;
    for (int e = 0; e < NEXP; ++e) {
      p[e] *= inv;
      gate_out[n * NEXP + e] = acc[e];  // raw logits
      rout_out[n * NEXP + e] = p[e];    // softmax routing
    }
    // top-2, first-index wins on ties (matches jax top_k)
    int i1 = 0;
    for (int e = 1; e < NEXP; ++e) if (p[e] > p[i1]) i1 = e;
    int i2 = (i1 == 0) ? 1 : 0;
    for (int e = 0; e < NEXP; ++e) if (e != i1 && p[e] > p[i2]) i2 = e;
    float sw = p[i1] + p[i2];
    float w1 = p[i1] / sw, w2 = p[i2] / sw;
    for (int e = 0; e < NEXP; ++e)
      combined[n * NEXP + e] = (e == i1) ? w1 : ((e == i2) ? w2 : 0.f);
    int pos1 = atomicAdd(&counts[i1], 1);
    lists[i1 * NTOK + pos1] = n;
    int pos2 = atomicAdd(&counts[i2], 1);
    lists[i2 * NTOK + pos2] = n;
  }
}

// ---------------- fused gathered expert FFN (WMMA bf16) ----------------
// Block: 256 threads (8 waves). One block = one expert, one 16-token M tile.
// Each wave owns 128 output columns (8 WMMA C tiles, 64 fp32 VGPRs).
// Loop H in chunks of 256: each wave computes its 32-wide h slice
// (silu(x@W1ᵉᵀ) * (x@W3ᵉᵀ)) with WMMA, shares it via LDS, then all waves
// WMMA it against W2ᵉ into the output accumulators.
// The x tile is gathered memory->LDS with GLOBAL_LOAD_ASYNC_TO_LDS_B128
// (ASYNCcnt-tracked, no VGPR round-trip), settled with s_wait_asynccnt.

__global__ void __launch_bounds__(256) moe_ffn_kernel(
    const __bf16* __restrict__ xbf, const __bf16* __restrict__ W1bf,
    const __bf16* __restrict__ W2bf, const __bf16* __restrict__ W3bf,
    const float* __restrict__ combined, const int* __restrict__ counts,
    const int* __restrict__ lists, float* __restrict__ out_final,
    float* __restrict__ cls_contrib) {
  int e = blockIdx.x >> 8;     // 256 M-tiles per expert
  int tile = blockIdx.x & 255;
  int count = counts[e];
  if (tile * 16 >= count) return;  // uniform early exit for inactive tiles

  __shared__ __align__(16) __bf16 xs[16 * 1032];  // 16 tokens x D, padded row
  __shared__ __align__(16) __bf16 hs[16 * 264];   // 16 tokens x 256 h chunk

  int tid = threadIdx.x;
  const int* lst = lists + e * NTOK;

  // Async gather of this tile's 16 bf16 token rows into LDS.
  // 2048 16-byte fragments; each lane pushes 8 async B128 transfers whose
  // LDS destination comes from the lane VGPR (per-lane scatter into xs).
#pragma unroll
  for (int k = 0; k < 8; ++k) {
    int i = tid + k * 256;
    int r = i >> 7, seg = i & 127;
    int gRow = tile * 16 + r;
    int tok = (gRow < count) ? lst[gRow] : 0;
    unsigned lds_off = (unsigned)(size_t)(xs + r * 1032 + seg * 8);
    unsigned long long gaddr =
        (unsigned long long)(size_t)(xbf + (size_t)tok * DDIM + seg * 8);
    asm volatile("global_load_async_to_lds_b128 %0, %1, off"
                 :
                 : "v"(lds_off), "v"(gaddr)
                 : "memory");
  }
  asm volatile("s_wait_asynccnt 0x0" ::: "memory");
  __syncthreads();

  int lane = tid & 31, wv = tid >> 5;
  int lrow = lane & 15;
  int koff = (lane >> 4) << 3;  // K sub-offset per bf16 WMMA lane striping
  int hi8 = (lane >> 4) << 3;   // C/D row offset for lanes 16..31

  v8f acc[8];
#pragma unroll
  for (int nt = 0; nt < 8; ++nt) acc[nt] = vzero8();

  const __bf16* xrow = xs + lrow * 1032 + koff;

  for (int hb = 0; hb < 8; ++hb) {
    // Warm up the W2 panel this wave will consume in stage 2 of this chunk
    // (emits global_prefetch_b8; GEMM1 below gives it ~128 WMMAs of cover).
    __builtin_prefetch(
        W2bf + ((size_t)e * DDIM + wv * 128 + lrow) * HDIM + hb * 256, 0, 1);

    // ---- stage 1: h[16 x 32] for this wave's slice of the H chunk ----
    int j0 = hb * 256 + wv * 32;
    const __bf16* w1a = W1bf + ((size_t)e * HDIM + j0 + lrow) * DDIM + koff;
    const __bf16* w1b = W1bf + ((size_t)e * HDIM + j0 + 16 + lrow) * DDIM + koff;
    const __bf16* w3a = W3bf + ((size_t)e * HDIM + j0 + lrow) * DDIM + koff;
    const __bf16* w3b = W3bf + ((size_t)e * HDIM + j0 + 16 + lrow) * DDIM + koff;
    v8f c1a = vzero8(), c1b = vzero8(), c3a = vzero8(), c3b = vzero8();
    for (int kd = 0; kd < DDIM; kd += 32) {
      v16bf a = ld16bf(xrow + kd);
      c1a = wmma_bf16(a, ld16bf(w1a + kd), c1a);
      c1b = wmma_bf16(a, ld16bf(w1b + kd), c1b);
      c3a = wmma_bf16(a, ld16bf(w3a + kd), c3a);
      c3b = wmma_bf16(a, ld16bf(w3b + kd), c3b);
    }
    // silu(z1) * z3 elementwise in the C register layout, write bf16 to LDS
#pragma unroll
    for (int i = 0; i < 8; ++i) {
      float z = c1a[i];
      float h = (z / (1.f + __expf(-z))) * c3a[i];
      hs[(i + hi8) * 264 + wv * 32 + lrow] = (__bf16)h;
      z = c1b[i];
      h = (z / (1.f + __expf(-z))) * c3b[i];
      hs[(i + hi8) * 264 + wv * 32 + 16 + lrow] = (__bf16)h;
    }
    __syncthreads();

    // ---- stage 2: acc[16 x 128] += h[16 x 256] @ W2ᵉ[128 x 256]ᵀ ----
    const __bf16* hrow = hs + lrow * 264 + koff;
    for (int k2 = 0; k2 < 256; k2 += 32) {
      v16bf a = ld16bf(hrow + k2);
#pragma unroll
      for (int nt = 0; nt < 8; ++nt) {
        int dout = wv * 128 + nt * 16 + lrow;
        v16bf b = ld16bf(W2bf + ((size_t)e * DDIM + dout) * HDIM +
                         hb * 256 + k2 + koff);
        acc[nt] = wmma_bf16(a, b, acc[nt]);
      }
    }
    __syncthreads();  // hs reused next iteration
  }

  // ---- weighted scatter: final += contrib * combined; CLS side-channel ----
  int toks[8];
  float wc8[8];
#pragma unroll
  for (int i = 0; i < 8; ++i) {
    int gRow = tile * 16 + i + hi8;
    bool v = gRow < count;
    toks[i] = v ? lst[gRow] : -1;
    wc8[i] = v ? combined[toks[i] * NEXP + e] : 0.f;
  }
#pragma unroll
  for (int nt = 0; nt < 8; ++nt) {
    int col = wv * 128 + nt * 16 + lrow;
#pragma unroll
    for (int i = 0; i < 8; ++i) {
      if (toks[i] < 0) continue;
      float v = acc[nt][i] * wc8[i];
      atomicAdd(out_final + (size_t)toks[i] * DDIM + col, v);
      if ((toks[i] & (SS - 1)) == 0) {            // CLS token (s == 0)
        int b = toks[i] >> 10;
        cls_contrib[((e * BB + b) << 10) + col] = v;  // single writer
      }
    }
  }
}

// ---------------- cumulative CLS classifier ----------------
// expert_logits[e][b] = sum_{e'<=e} dot(cls_contrib[e'][b], Wc) + bc

__global__ void __launch_bounds__(256) cls_kernel(
    const float* __restrict__ cls_contrib, const float* __restrict__ Wc,
    const float* __restrict__ bc, float* __restrict__ elog) {
  __shared__ float dots[NEXP][BB];
  int tid = threadIdx.x, lane = tid & 31, wv = tid >> 5;  // wave = expert
  for (int b = 0; b < BB; ++b) {
    float s = 0.f;
    const float* c = cls_contrib + ((wv * BB + b) << 10);
    for (int d = lane; d < DDIM; d += 32) s += c[d] * Wc[d];
#pragma unroll
    for (int off = 16; off > 0; off >>= 1) s += __shfl_xor(s, off, 32);
    if (lane == 0) dots[wv][b] = s;
  }
  __syncthreads();
  if (tid < NEXP * BB) {
    int e = tid >> 2, b = tid & 3;
    float t = bc[0];
    for (int e2 = 0; e2 <= e; ++e2) t += dots[e2][b];
    elog[e * BB + b] = t;
  }
}

// ---------------- launcher ----------------

extern "C" void kernel_launch(void* const* d_in, const int* in_sizes, int n_in,
                              void* d_out, int out_size, void* d_ws,
                              size_t ws_size, hipStream_t stream) {
  (void)in_sizes; (void)n_in; (void)out_size; (void)ws_size;
  const float* x  = (const float*)d_in[0];
  // d_in[1] = tgt_pad: unused by the reference
  const float* W1 = (const float*)d_in[2];
  const float* W2 = (const float*)d_in[3];
  const float* W3 = (const float*)d_in[4];
  const float* Wg = (const float*)d_in[5];
  const float* Wc = (const float*)d_in[6];
  const float* bc = (const float*)d_in[7];

  float* outf = (float*)d_out;
  float* out_final = outf;                            // 4096*1024
  float* out_gate  = outf + 4194304;                  // 4096*8
  float* out_elog  = outf + 4194304 + 32768;          // 8*4
  float* out_rout  = outf + 4194304 + 32768 + 32;     // 4096*8

  // workspace layout (~104.4 MB)
  char* ws = (char*)d_ws;
  __bf16* xbf     = (__bf16*)(ws);                    //  8 MB
  __bf16* W1bf    = (__bf16*)(ws + 8388608ull);       // 32 MB
  __bf16* W2bf    = (__bf16*)(ws + 41943040ull);      // 32 MB
  __bf16* W3bf    = (__bf16*)(ws + 75497472ull);      // 32 MB
  float*  combined = (float*)(ws + 109051904ull);     // 128 KB
  int*    lists    = (int*)  (ws + 109182976ull);     // 128 KB
  int*    counts   = (int*)  (ws + 109314048ull);     // 64 B
  float*  clsc     = (float*)(ws + 109314112ull);     // 128 KB

  zero_f32<<<4096, 256, 0, stream>>>(out_final, 1048576);
  zero_f32<<<32, 256, 0, stream>>>(clsc, 8192);
  zero_f32<<<1, 32, 0, stream>>>((float*)counts, 2);

  cvt_bf16<<<4096, 256, 0, stream>>>(x, xbf, 1048576);
  cvt_bf16<<<16384, 256, 0, stream>>>(W1, W1bf, 4194304);
  cvt_bf16<<<16384, 256, 0, stream>>>(W2, W2bf, 4194304);
  cvt_bf16<<<16384, 256, 0, stream>>>(W3, W3bf, 4194304);

  gate_kernel<<<512, 256, 0, stream>>>(x, Wg, out_gate, out_rout, combined,
                                       counts, lists);
  moe_ffn_kernel<<<NEXP * 256, 256, 0, stream>>>(xbf, W1bf, W2bf, W3bf,
                                                 combined, counts, lists,
                                                 out_final, clsc);
  cls_kernel<<<1, 256, 0, stream>>>(clsc, Wc, bc, out_elog);
}
